// QuantizedDenseWithBN_27745488732699
// MI455X (gfx1250) — 8-bit matrix operand order repaired and checked by reading, compiled, not run on hardware
//
#include <hip/hip_runtime.h>
#include <math.h>


typedef float        v2f   __attribute__((ext_vector_type(2)));
typedef float        v8f   __attribute__((ext_vector_type(8)));
typedef int          v8i   __attribute__((ext_vector_type(8)));
typedef int          i32x4 __attribute__((ext_vector_type(4)));
typedef unsigned int u32x4 __attribute__((ext_vector_type(4)));

#define B_ROWS  32768
#define K_DIM   512
#define N_UN    512
#define EPSV    1e-5f

#define XS_STRIDE 514

#define OFF_MMP ((size_t)0)
#define OFF_PAR ((size_t)(8  << 10))
#define OFF_BF  ((size_t)(16 << 10))
#define OFF_CSR ((size_t)(20 << 10))
#define OFF_RCM ((size_t)(24 << 10))
#define OFF_RQ  ((size_t)(512 << 10))
#define OFF_M1  ((size_t)(1 << 20))
#define OFF_M2  ((size_t)(5 << 20))
#define OFF_Q   ((size_t)(9 << 20))

__global__ void k_minmax_part(const float* __restrict__ x, float* __restrict__ mmp) {
    __shared__ float smn[256], smx[256];
    const int tid = threadIdx.x;
    const float4* x4 = (const float4*)x;
    const size_t base = (size_t)blockIdx.x * 4096;
    float mn = 3.4e38f, mx = -3.4e38f;
#pragma unroll
    for (int i = 0; i < 16; ++i) {
        float4 v = x4[base + i * 256 + tid];
        mn = fminf(mn, fminf(fminf(v.x, v.y), fminf(v.z, v.w)));
        mx = fmaxf(mx, fmaxf(fmaxf(v.x, v.y), fmaxf(v.z, v.w)));
    }
    smn[tid] = mn; smx[tid] = mx;
    __syncthreads();
    for (int s = 128; s > 0; s >>= 1) {
        if (tid < s) {
            smn[tid] = fminf(smn[tid], smn[tid + s]);
            smx[tid] = fmaxf(smx[tid], smx[tid + s]);
        }
        __syncthreads();
    }
    if (tid == 0) { mmp[blockIdx.x * 2] = smn[0]; mmp[blockIdx.x * 2 + 1] = smx[0]; }
}

__global__ void k_minmax_final(const float* __restrict__ mmp, float* __restrict__ par) {
    __shared__ float smn[256], smx[256];
    const int tid = threadIdx.x;
    float mn = 3.4e38f, mx = -3.4e38f;
    for (int i = tid; i < 1024; i += 256) {
        mn = fminf(mn, mmp[i * 2]);
        mx = fmaxf(mx, mmp[i * 2 + 1]);
    }
    smn[tid] = mn; smx[tid] = mx;
    __syncthreads();
    for (int s = 128; s > 0; s >>= 1) {
        if (tid < s) {
            smn[tid] = fminf(smn[tid], smn[tid + s]);
            smx[tid] = fmaxf(smx[tid], smx[tid + s]);
        }
        __syncthreads();
    }
    if (tid == 0) {
        par[0] = smn[0];
        par[1] = (smx[0] - smn[0]) / 255.0f;
    }
}

#if defined(__gfx1250__)
__device__ __forceinline__ void tdm_load_x_slab(const float* gptr, void* lds_ptr) {
    const unsigned long long ga = (unsigned long long)(size_t)gptr;
    const unsigned int lofs = (unsigned int)(size_t)lds_ptr;
    u32x4 g0;
    g0[0] = 1u;
    g0[1] = lofs;
    g0[2] = (unsigned int)(ga & 0xFFFFFFFFu);
    g0[3] = (unsigned int)((ga >> 32) & 0x01FFFFFFu)
          | (2u << 30);
    v8i g1;
    g1[0] = (int)((2u << 16)
                | (1u << 20)
                | (7u << 22));
    g1[1] = (int)(512u << 16);
    g1[2] = (int)(16u << 16);
    g1[3] = (int)(512u << 16);
    g1[4] = 16;
    g1[5] = 512;
    g1[6] = 0;
    g1[7] = 0;
    const i32x4 z4 = {0, 0, 0, 0};
#if defined(__clang_major__) && (__clang_major__ >= 23)
    const v8i z8 = {0, 0, 0, 0, 0, 0, 0, 0};
    __builtin_amdgcn_tensor_load_to_lds(g0, g1, z4, z4, z8, 0);
#else
    __builtin_amdgcn_tensor_load_to_lds(g0, g1, z4, z4, 0);
#endif
}
#endif

__device__ __forceinline__ int xs_idx(int r, int k) {
    return r * XS_STRIDE + k + (k >> 8);
}

__global__ void k_moments(const float* __restrict__ x, const float* __restrict__ W,
                          float* __restrict__ m1, float* __restrict__ m2) {
    __shared__ float xs[16 * XS_STRIDE + 8];
    const int slab = blockIdx.x;
    const int row0 = slab * 16;
    const int tid  = threadIdx.x;

#if defined(__gfx1250__)
    if (tid < 32) {
        tdm_load_x_slab(x + (size_t)row0 * K_DIM, (void*)xs);
        __builtin_amdgcn_s_wait_tensorcnt(0);
    }
#else
    for (int i = tid; i < 16 * 512; i += 256) {
        const int r = i >> 9, k = i & 511;
        xs[xs_idx(r, k)] = x[(size_t)(row0 + r) * K_DIM + k];
    }
#endif
    __syncthreads();

    const int wave = tid >> 5, lane = tid & 31;
    const int half = lane >> 4, l15 = lane & 15;
    v8f acc[4] = {};

    for (int k = 0; k < K_DIM; k += 4) {
        const int kk = k + half * 2;
        const int ai = xs_idx(l15, kk);
        v2f a;
        a.x = xs[ai];
        a.y = xs[ai + 1];
#pragma unroll
        for (int t = 0; t < 4; ++t) {
            const int col0 = (wave + 8 * t) * 16;
            v2f b;
            b.x = W[(size_t)kk * N_UN + col0 + l15];
            b.y = W[(size_t)(kk + 1) * N_UN + col0 + l15];
            acc[t] = __builtin_amdgcn_wmma_f32_16x16x4_f32(
                false, a, false, b, (short)0, acc[t], false, false);
        }
    }
#pragma unroll
    for (int t = 0; t < 4; ++t) {
        float s1 = 0.0f, s2 = 0.0f;
#pragma unroll
        for (int v = 0; v < 8; ++v) { const float e = acc[t][v]; s1 += e; s2 += e * e; }
        s1 += __shfl_xor(s1, 16, 32);
        s2 += __shfl_xor(s2, 16, 32);
        if (half == 0) {
            const int col = (wave + 8 * t) * 16 + l15;
            m1[(size_t)slab * N_UN + col] = s1;
            m2[(size_t)slab * N_UN + col] = s2;
        }
    }
}

__global__ void k_prep(const float* __restrict__ W, const float* __restrict__ scale,
                       const float* __restrict__ offs, const float* __restrict__ m1,
                       const float* __restrict__ m2, float* __restrict__ par,
                       float* __restrict__ bfold, float* __restrict__ csr,
                       unsigned char* __restrict__ rcm) {
    __shared__ float red0[512], red1[512];
    __shared__ float s_wmn, s_wstep;
    const int j = threadIdx.x;

    float s1 = 0.0f, s2 = 0.0f;
    for (int s = 0; s < 2048; ++s) {
        s1 += m1[(size_t)s * N_UN + j];
        s2 += m2[(size_t)s * N_UN + j];
    }
    const float mean = s1 * (1.0f / 32768.0f);
    const float var  = s2 * (1.0f / 32768.0f) - mean * mean;
    const float inv  = 1.0f / sqrtf(var + EPSV);
    const float sj   = scale[j] * inv;
    bfold[j] = offs[j] - scale[j] * mean * inv;

    float wmn = 3.4e38f, wmx = -3.4e38f;
    for (int k = 0; k < K_DIM; ++k) {
        const float wf = sj * W[(size_t)k * N_UN + j];
        wmn = fminf(wmn, wf); wmx = fmaxf(wmx, wf);
    }
    red0[j] = wmn; red1[j] = wmx;
    __syncthreads();
    for (int s = 256; s > 0; s >>= 1) {
        if (j < s) {
            red0[j] = fminf(red0[j], red0[j + s]);
            red1[j] = fmaxf(red1[j], red1[j + s]);
        }
        __syncthreads();
    }
    if (j == 0) {
        s_wmn   = red0[0];
        s_wstep = (red1[0] - red0[0]) / 255.0f;
        par[2] = s_wmn; par[3] = s_wstep;
    }
    __syncthreads();
    const float wmn0 = s_wmn, wst = s_wstep;

    float cs = 0.0f;
    for (int k = 0; k < K_DIM; ++k) {
        const float wf = sj * W[(size_t)k * N_UN + j];
        const float r  = rintf((wf - wmn0) / wst);
        rcm[(size_t)j * K_DIM + k] = (unsigned char)(int)r;
        cs += r;
    }
    csr[j] = cs;
}

__global__ void k_quant_x(const float* __restrict__ x, const float* __restrict__ par,
                          unsigned char* __restrict__ Q, float* __restrict__ rowsum) {
    const int wave = threadIdx.x >> 5, lane = threadIdx.x & 31;
    const int row  = blockIdx.x * 8 + wave;
    const float qmn = par[0], qstep = par[1];
    const float4* xr = (const float4*)(x + (size_t)row * K_DIM);
    uchar4*       qr = (uchar4*)(Q + (size_t)row * K_DIM);
    float s = 0.0f;
#pragma unroll
    for (int i = 0; i < 4; ++i) {
        const int c4 = lane + 32 * i;
        const float4 v = xr[c4];
        const float q0 = rintf((v.x - qmn) / qstep);
        const float q1 = rintf((v.y - qmn) / qstep);
        const float q2 = rintf((v.z - qmn) / qstep);
        const float q3 = rintf((v.w - qmn) / qstep);
        uchar4 u;
        u.x = (unsigned char)(int)q0; u.y = (unsigned char)(int)q1;
        u.z = (unsigned char)(int)q2; u.w = (unsigned char)(int)q3;
        qr[c4] = u;
        s += q0 + q1 + q2 + q3;
    }
#pragma unroll
    for (int m = 16; m > 0; m >>= 1) s += __shfl_xor(s, m, 32);
    if (lane == 0) rowsum[row] = s;
}

__global__ void k_gemm_iu8(const unsigned char* __restrict__ Q,
                           const unsigned char* __restrict__ Rcm,
                           const float* __restrict__ par,
                           const float* __restrict__ bfold,
                           const float* __restrict__ csr,
                           const float* __restrict__ rowsum,
                           float* __restrict__ out) {
    const int wave = threadIdx.x >> 5, lane = threadIdx.x & 31;
    const int nt   = blockIdx.x & 31;
    const int mblk = blockIdx.x >> 5;
    const int row0 = mblk * 128 + wave * 16;
    const int col0 = nt * 16;
    const int half = lane >> 4, l15 = lane & 15;

    const unsigned int* Qd = (const unsigned int*)Q;
    const unsigned int* Rd = (const unsigned int*)Rcm;
    const int arow = row0 + l15;
    const int bcol = col0 + l15;

    const uint2* Qa = (const uint2*)(Qd + (size_t)arow * 128);
    const uint4* Rb = (const uint4*)(Rd + (size_t)bcol * 128);

    v8i acc = {0, 0, 0, 0, 0, 0, 0, 0};
#pragma unroll
    for (int ks = 0; ks < 8; ++ks) {
        const int aoff = ks * 8 + half * 2;
        const uint2 a01 = Qa[aoff + 0];
        const uint2 a23 = Qa[aoff + 1];
        const uint2 a45 = Qa[aoff + 4];
        const uint2 a67 = Qa[aoff + 5];
        const int boff = ks * 4 + half;
        const uint4 b03 = Rb[boff + 0];
        const uint4 b47 = Rb[boff + 2];
        v8i a, b;
        a[0] = (int)a01.x; a[1] = (int)a01.y; a[2] = (int)a23.x; a[3] = (int)a23.y;
        a[4] = (int)a45.x; a[5] = (int)a45.y; a[6] = (int)a67.x; a[7] = (int)a67.y;
        b[0] = (int)b03.x; b[1] = (int)b03.y; b[2] = (int)b03.z; b[3] = (int)b03.w;
        b[4] = (int)b47.x; b[5] = (int)b47.y; b[6] = (int)b47.z; b[7] = (int)b47.w;
        acc = __builtin_amdgcn_wmma_i32_16x16x64_iu8(false, a, false, b, acc,
                                                     false, false);
    }

    const float qmn = par[0], qstep = par[1], wmn = par[2], wstep = par[3];
    const float s_qw = qstep * wstep;
    const float s_q  = qstep * wmn;
    const float cconst = qmn * wstep * csr[bcol] + 512.0f * qmn * wmn + bfold[bcol];
#pragma unroll
    for (int v = 0; v < 8; ++v) {
        const int row = row0 + v + half * 8;
        out[(size_t)row * N_UN + bcol] =
            s_qw * (float)acc[v] + s_q * rowsum[row] + cconst;
    }
}

extern "C" void kernel_launch(void* const* d_in, const int* in_sizes, int n_in,
                              void* d_out, int out_size, void* d_ws, size_t ws_size,
                              hipStream_t stream) {
    (void)in_sizes; (void)n_in; (void)out_size; (void)ws_size;
    const float* x     = (const float*)d_in[0];
    const float* W     = (const float*)d_in[1];
    const float* scale = (const float*)d_in[2];
    const float* offs  = (const float*)d_in[3];
    float* out = (float*)d_out;

    char* ws = (char*)d_ws;
    float* mmp = (float*)(ws + OFF_MMP);
    float* par = (float*)(ws + OFF_PAR);
    float* bf  = (float*)(ws + OFF_BF);
    float* csr = (float*)(ws + OFF_CSR);
    unsigned char* rcm = (unsigned char*)(ws + OFF_RCM);
    float* rq  = (float*)(ws + OFF_RQ);
    float* m1  = (float*)(ws + OFF_M1);
    float* m2  = (float*)(ws + OFF_M2);
    unsigned char* Q = (unsigned char*)(ws + OFF_Q);

    k_minmax_part <<<1024, 256, 0, stream>>>(x, mmp);
    k_minmax_final<<<1,    256, 0, stream>>>(mmp, par);
    k_moments     <<<2048, 256, 0, stream>>>(x, W, m1, m2);
    k_prep        <<<1,    512, 0, stream>>>(W, scale, offs, m1, m2, par, bf, csr, rcm);
    k_quant_x     <<<4096, 256, 0, stream>>>(x, par, Q, rq);
    k_gemm_iu8    <<<8192, 256, 0, stream>>>(Q, rcm, par, bf, csr, rq, out);
}
